// TinyTransformerBlock_1915555414404
// MI455X (gfx1250) — matrix operand order repaired and checked by reading, not compiled, not run on hardware
//
#include <hip/hip_runtime.h>
#include <hip/hip_fp16.h>

typedef __attribute__((ext_vector_type(16))) _Float16 v16h;
typedef __attribute__((ext_vector_type(8)))  _Float16 v8h;
typedef __attribute__((ext_vector_type(4)))  _Float16 v4h;
typedef __attribute__((ext_vector_type(8)))  float    v8f;

#define DIMD   1024
#define FFD    4096
#define SEQ    2048
#define NBATCH 4
#define NTOK   (NBATCH*SEQ)

// ---------------------------------------------------------------------------
// LayerNorm (row = 1024 fp32) -> f16 output
// ---------------------------------------------------------------------------
__global__ __launch_bounds__(256) void ln_f16_kernel(
    const float* __restrict__ x, const float* __restrict__ w,
    const float* __restrict__ b, _Float16* __restrict__ out)
{
    const int row = blockIdx.x;
    const int tid = threadIdx.x;
    const float4 xv = *reinterpret_cast<const float4*>(x + (size_t)row * DIMD + tid * 4);
    __shared__ float s1[256], s2[256];
    s1[tid] = xv.x + xv.y + xv.z + xv.w;
    s2[tid] = xv.x*xv.x + xv.y*xv.y + xv.z*xv.z + xv.w*xv.w;
    for (int st = 128; st > 0; st >>= 1) {
        __syncthreads();
        if (tid < st) { s1[tid] += s1[tid + st]; s2[tid] += s2[tid + st]; }
    }
    __syncthreads();
    const float mean = s1[0] * (1.0f / DIMD);
    const float var  = s2[0] * (1.0f / DIMD) - mean * mean;
    const float rs   = rsqrtf(var + 1e-5f);
    const float4 wv = *reinterpret_cast<const float4*>(w + tid * 4);
    const float4 bv = *reinterpret_cast<const float4*>(b + tid * 4);
    v4h o;
    o[0] = (_Float16)((xv.x - mean) * rs * wv.x + bv.x);
    o[1] = (_Float16)((xv.y - mean) * rs * wv.y + bv.y);
    o[2] = (_Float16)((xv.z - mean) * rs * wv.z + bv.z);
    o[3] = (_Float16)((xv.w - mean) * rs * wv.w + bv.w);
    *reinterpret_cast<v4h*>(out + (size_t)row * DIMD + tid * 4) = o;
}

// ---------------------------------------------------------------------------
// Row softmax over 2048 fp32 -> f16 probs
// ---------------------------------------------------------------------------
__global__ __launch_bounds__(256) void softmax_f16_kernel(
    const float* __restrict__ sc, _Float16* __restrict__ p)
{
    const int row = blockIdx.x;
    const int tid = threadIdx.x;
    const float* r = sc + (size_t)row * SEQ + tid * 8;
    float v[8];
#pragma unroll
    for (int i = 0; i < 8; i++) v[i] = r[i];
    __shared__ float red[256];
    float m = v[0];
#pragma unroll
    for (int i = 1; i < 8; i++) m = fmaxf(m, v[i]);
    red[tid] = m;
    for (int st = 128; st > 0; st >>= 1) {
        __syncthreads();
        if (tid < st) red[tid] = fmaxf(red[tid], red[tid + st]);
    }
    __syncthreads();
    m = red[0];
    __syncthreads();
    float s = 0.0f;
#pragma unroll
    for (int i = 0; i < 8; i++) { v[i] = __expf(v[i] - m); s += v[i]; }
    red[tid] = s;
    for (int st = 128; st > 0; st >>= 1) {
        __syncthreads();
        if (tid < st) red[tid] += red[tid + st];
    }
    __syncthreads();
    const float inv = 1.0f / red[0];
    v8h o;
#pragma unroll
    for (int i = 0; i < 8; i++) o[i] = (_Float16)(v[i] * inv);
    *reinterpret_cast<v8h*>(p + (size_t)row * SEQ + tid * 8) = o;
}

// ---------------------------------------------------------------------------
// Transpose+convert fp32 W[K][N] -> f16 Wt[N][K] (K-contiguous B layout)
// ---------------------------------------------------------------------------
__global__ __launch_bounds__(256) void transpose_convert_kernel(
    const float* __restrict__ W, _Float16* __restrict__ Wt, int K, int Nn)
{
    __shared__ float tile[32][33];
    const int gn = blockIdx.x * 32, gk = blockIdx.y * 32;
    const int tx = threadIdx.x, ty = threadIdx.y;   // block (32,8)
#pragma unroll
    for (int i = 0; i < 4; i++) {
        const int k = ty + i * 8;
        tile[k][tx] = W[(size_t)(gk + k) * Nn + gn + tx];
    }
    __syncthreads();
#pragma unroll
    for (int i = 0; i < 4; i++) {
        const int n = ty + i * 8;
        Wt[(size_t)(gn + n) * K + gk + tx] = (_Float16)tile[tx][n];
    }
}

// ---------------------------------------------------------------------------
// Generic f16 WMMA GEMM:  out = scale*(A @ B) + bias [+ residual] [ReLU]
//   A  : f16, row-major M x K, leading dim lda
//   Bt : f16, "transposed" layout: row n holds K contiguous values (ldb)
//   Each wave computes a 16x64 strip (4 accumulators, shared A fragment).
//   Output: fp32 (Cf) and/or f16 (Ch); transLd>0 stores f16 transposed.
// ---------------------------------------------------------------------------
__global__ __launch_bounds__(256) void wmma_gemm_kernel(
    const _Float16* __restrict__ A, const _Float16* __restrict__ Bt,
    const float* __restrict__ bias, const float* __restrict__ resid,
    float* __restrict__ Cf, _Float16* __restrict__ Ch,
    int M, int Nn, int K, int lda, int ldb, int ldc,
    float scale, int relu, int transLd)
{
    const int tid  = threadIdx.x;
    const int lane = tid & 31;
    const int wave = tid >> 5;
    const int rowTile = blockIdx.y * 2 + (wave >> 2);
    const int nStrip  = blockIdx.x * 4 + (wave & 3);
    const int rowBase = rowTile * 16;
    const int colBase = nStrip * 64;

    // A fragment addressing (contiguous k order, not the ISA table order; B uses the same):
    //   lanes 0-15: K 0-15 of row (lane&15); lanes 16-31: K 16-31
    const int arow = rowBase + (lane & 15);
    const int aoff = (lane & 16) ? 16 : 0;
    // B fragment addressing (16-bit B 32x16): lanes 0-15 -> K 0-15,
    //   lanes 16-31 -> K 16-31; column = colBase + t*16 + (lane&15)
    const int bcol = colBase + (lane & 15);
    const int boff = (lane & 16) ? 16 : 0;

    v8f acc0 = {}, acc1 = {}, acc2 = {}, acc3 = {};
    const _Float16* aRow = A + (size_t)arow * lda;

    for (int kb = 0; kb < K; kb += 32) {
        v16h a;
        reinterpret_cast<uint4*>(&a)[0] = *reinterpret_cast<const uint4*>(aRow + kb + aoff);
        reinterpret_cast<uint4*>(&a)[1] = *reinterpret_cast<const uint4*>(aRow + kb + 8 + aoff);

        v16h b0, b1, b2, b3;
        {
            const _Float16* p = Bt + (size_t)(bcol +  0) * ldb + kb + boff;
            reinterpret_cast<uint4*>(&b0)[0] = reinterpret_cast<const uint4*>(p)[0];
            reinterpret_cast<uint4*>(&b0)[1] = reinterpret_cast<const uint4*>(p)[1];
        }
        {
            const _Float16* p = Bt + (size_t)(bcol + 16) * ldb + kb + boff;
            reinterpret_cast<uint4*>(&b1)[0] = reinterpret_cast<const uint4*>(p)[0];
            reinterpret_cast<uint4*>(&b1)[1] = reinterpret_cast<const uint4*>(p)[1];
        }
        {
            const _Float16* p = Bt + (size_t)(bcol + 32) * ldb + kb + boff;
            reinterpret_cast<uint4*>(&b2)[0] = reinterpret_cast<const uint4*>(p)[0];
            reinterpret_cast<uint4*>(&b2)[1] = reinterpret_cast<const uint4*>(p)[1];
        }
        {
            const _Float16* p = Bt + (size_t)(bcol + 48) * ldb + kb + boff;
            reinterpret_cast<uint4*>(&b3)[0] = reinterpret_cast<const uint4*>(p)[0];
            reinterpret_cast<uint4*>(&b3)[1] = reinterpret_cast<const uint4*>(p)[1];
        }
        acc0 = __builtin_amdgcn_wmma_f32_16x16x32_f16(false, a, false, b0, (short)0, acc0, false, false);
        acc1 = __builtin_amdgcn_wmma_f32_16x16x32_f16(false, a, false, b1, (short)0, acc1, false, false);
        acc2 = __builtin_amdgcn_wmma_f32_16x16x32_f16(false, a, false, b2, (short)0, acc2, false, false);
        acc3 = __builtin_amdgcn_wmma_f32_16x16x32_f16(false, a, false, b3, (short)0, acc3, false, false);
    }

    // Epilogue. C/D f32 layout: lane covers col = (lane&15); lanes 0-15 hold
    // rows 0-7 in VGPRs 0-7, lanes 16-31 hold rows 8-15.
    const int rbase = rowBase + ((lane & 16) ? 8 : 0);
#pragma unroll
    for (int t = 0; t < 4; t++) {
        const v8f c = (t == 0) ? acc0 : (t == 1) ? acc1 : (t == 2) ? acc2 : acc3;
        const int col = colBase + t * 16 + (lane & 15);
        const float bsv = bias ? bias[col] : 0.0f;
        if (transLd > 0) {
            v8h hv;
#pragma unroll
            for (int r = 0; r < 8; r++) {
                float v = c[r] * scale + bsv;
                if (resid) v += resid[(size_t)(rbase + r) * ldc + col];
                if (relu)  v = fmaxf(v, 0.0f);
                hv[r] = (_Float16)v;
            }
            *reinterpret_cast<uint4*>(Ch + (size_t)col * transLd + rbase) =
                *reinterpret_cast<uint4*>(&hv);
        } else {
#pragma unroll
            for (int r = 0; r < 8; r++) {
                float v = c[r] * scale + bsv;
                const size_t idx = (size_t)(rbase + r) * ldc + col;
                if (resid) v += resid[idx];
                if (relu)  v = fmaxf(v, 0.0f);
                if (Cf) Cf[idx] = v;
                if (Ch) Ch[idx] = (_Float16)v;
            }
        }
    }
}

// ---------------------------------------------------------------------------
extern "C" void kernel_launch(void* const* d_in, const int* in_sizes, int n_in,
                              void* d_out, int out_size, void* d_ws, size_t ws_size,
                              hipStream_t stream)
{
    const float* x     = (const float*)d_in[0];
    const float* ln1_w = (const float*)d_in[1];
    const float* ln1_b = (const float*)d_in[2];
    const float* Wq    = (const float*)d_in[3];
    const float* bq    = (const float*)d_in[4];
    const float* Wk    = (const float*)d_in[5];
    const float* bk    = (const float*)d_in[6];
    const float* Wv    = (const float*)d_in[7];
    const float* bv    = (const float*)d_in[8];
    const float* Wo    = (const float*)d_in[9];
    const float* bo    = (const float*)d_in[10];
    const float* W1    = (const float*)d_in[11];
    const float* b1    = (const float*)d_in[12];
    const float* W2    = (const float*)d_in[13];
    const float* b2    = (const float*)d_in[14];
    const float* ln2_w = (const float*)d_in[15];
    const float* ln2_b = (const float*)d_in[16];
    float* out = (float*)d_out;

    char* wp = (char*)d_ws;
    auto alloc = [&](size_t bytes) -> void* {
        void* r = wp; wp += ((bytes + 255) / 256) * 256; return r;
    };
    _Float16* hh   = (_Float16*)alloc((size_t)NTOK * DIMD * 2);  // h / h2 (reused)
    _Float16* qh   = (_Float16*)alloc((size_t)NTOK * DIMD * 2);
    _Float16* kh   = (_Float16*)alloc((size_t)NTOK * DIMD * 2);
    _Float16* vt   = (_Float16*)alloc((size_t)NTOK * DIMD * 2);  // per-batch d x S
    _Float16* ctxh = (_Float16*)alloc((size_t)NTOK * DIMD * 2);
    _Float16* Wqt  = (_Float16*)alloc((size_t)DIMD * DIMD * 2);
    _Float16* Wkt  = (_Float16*)alloc((size_t)DIMD * DIMD * 2);
    _Float16* Wvt  = (_Float16*)alloc((size_t)DIMD * DIMD * 2);
    _Float16* Wot  = (_Float16*)alloc((size_t)DIMD * DIMD * 2);
    _Float16* W1t  = (_Float16*)alloc((size_t)DIMD * FFD * 2);
    _Float16* W2t  = (_Float16*)alloc((size_t)DIMD * FFD * 2);
    float*    scores = (float*)alloc((size_t)NBATCH * SEQ * SEQ * 4);
    _Float16* probs  = (_Float16*)alloc((size_t)NBATCH * SEQ * SEQ * 2);
    _Float16* f1h = (_Float16*)scores;   // alias: scores dead after softmax

    const dim3 tb(32, 8);
    // Weight convert+transpose (one pass, L2-resident afterwards)
    transpose_convert_kernel<<<dim3(DIMD/32, DIMD/32), tb, 0, stream>>>(Wq, Wqt, DIMD, DIMD);
    transpose_convert_kernel<<<dim3(DIMD/32, DIMD/32), tb, 0, stream>>>(Wk, Wkt, DIMD, DIMD);
    transpose_convert_kernel<<<dim3(DIMD/32, DIMD/32), tb, 0, stream>>>(Wv, Wvt, DIMD, DIMD);
    transpose_convert_kernel<<<dim3(DIMD/32, DIMD/32), tb, 0, stream>>>(Wo, Wot, DIMD, DIMD);
    transpose_convert_kernel<<<dim3(FFD/32,  DIMD/32), tb, 0, stream>>>(W1, W1t, DIMD, FFD);
    transpose_convert_kernel<<<dim3(DIMD/32, FFD/32),  tb, 0, stream>>>(W2, W2t, FFD, DIMD);

    // LN1
    ln_f16_kernel<<<NTOK, 256, 0, stream>>>(x, ln1_w, ln1_b, hh);

    // Q, K projections (f16 row-major out)
    wmma_gemm_kernel<<<dim3(DIMD/256, NTOK/32), 256, 0, stream>>>(
        hh, Wqt, bq, nullptr, nullptr, qh, NTOK, DIMD, DIMD, DIMD, DIMD, DIMD, 1.0f, 0, 0);
    wmma_gemm_kernel<<<dim3(DIMD/256, NTOK/32), 256, 0, stream>>>(
        hh, Wkt, bk, nullptr, nullptr, kh, NTOK, DIMD, DIMD, DIMD, DIMD, DIMD, 1.0f, 0, 0);
    // V projection, stored transposed per batch (d x S) for attn@V B-frags
    for (int b = 0; b < NBATCH; b++) {
        wmma_gemm_kernel<<<dim3(DIMD/256, SEQ/32), 256, 0, stream>>>(
            hh + (size_t)b * SEQ * DIMD, Wvt, bv, nullptr, nullptr,
            vt + (size_t)b * DIMD * SEQ, SEQ, DIMD, DIMD, DIMD, DIMD, DIMD, 1.0f, 0, SEQ);
    }
    // scores = (q @ k^T) / 32 ; kh row-major is exactly the Bt layout
    for (int b = 0; b < NBATCH; b++) {
        wmma_gemm_kernel<<<dim3(SEQ/256, SEQ/32), 256, 0, stream>>>(
            qh + (size_t)b * SEQ * DIMD, kh + (size_t)b * SEQ * DIMD,
            nullptr, nullptr, scores + (size_t)b * SEQ * SEQ, nullptr,
            SEQ, SEQ, DIMD, DIMD, DIMD, SEQ, 0.03125f, 0, 0);
    }
    // softmax -> f16 probs
    softmax_f16_kernel<<<NBATCH * SEQ, 256, 0, stream>>>(scores, probs);
    // ctx = probs @ v  (Bt = v^T, K-contiguous)
    for (int b = 0; b < NBATCH; b++) {
        wmma_gemm_kernel<<<dim3(DIMD/256, SEQ/32), 256, 0, stream>>>(
            probs + (size_t)b * SEQ * SEQ, vt + (size_t)b * DIMD * SEQ,
            nullptr, nullptr, nullptr, ctxh + (size_t)b * SEQ * DIMD,
            SEQ, DIMD, SEQ, SEQ, SEQ, DIMD, 1.0f, 0, 0);
    }
    // x1 = x + ctx @ Wo + bo  -> d_out (fp32)
    wmma_gemm_kernel<<<dim3(DIMD/256, NTOK/32), 256, 0, stream>>>(
        ctxh, Wot, bo, x, out, nullptr, NTOK, DIMD, DIMD, DIMD, DIMD, DIMD, 1.0f, 0, 0);
    // LN2 (reuse hh)
    ln_f16_kernel<<<NTOK, 256, 0, stream>>>(out, ln2_w, ln2_b, hh);
    // FFN1: relu(h2 @ W1 + b1) -> f16 (aliases scores buffer)
    wmma_gemm_kernel<<<dim3(FFD/256, NTOK/32), 256, 0, stream>>>(
        hh, W1t, b1, nullptr, nullptr, f1h, NTOK, FFD, DIMD, DIMD, DIMD, FFD, 1.0f, 1, 0);
    // FFN2 + bias + residual(x1) -> d_out
    wmma_gemm_kernel<<<dim3(DIMD/256, NTOK/32), 256, 0, stream>>>(
        f1h, W2t, b2, out, out, nullptr, NTOK, DIMD, FFD, FFD, FFD, DIMD, 1.0f, 0, 0);

    (void)in_sizes; (void)n_in; (void)out_size; (void)ws_size;
}
